// LIIF_6880537608891
// MI455X (gfx1250) — matrix operand order repaired and checked by reading, not compiled, not run on hardware
//
#include <hip/hip_runtime.h>

// ---------------------------------------------------------------------------
// LIIF forward for MI455X (gfx1250): f16 WMMA MLP, wave32.
// Round 2: 4-deep rotating register pipeline for B fragments so
// global_load_b128 stays 8-deep in flight across the whole layer.
// ---------------------------------------------------------------------------

typedef _Float16 v8h  __attribute__((ext_vector_type(8)));
typedef _Float16 v16h __attribute__((ext_vector_type(16)));
typedef float    v8f  __attribute__((ext_vector_type(8)));

#define LIIF_B   4
#define LIIF_C   64
#define LIIF_H   64
#define LIIF_W   64
#define LIIF_Q   65536
#define STR0     608              // buf0 row stride (halves), 580 padded to 19*32
#define STR1     272              // buf1 row stride (halves, 256 + pad)

// per-wave LDS region layout (bytes)
#define OFF_BUF0  0               // 16*608*2 = 19456
#define OFF_BUF1  19456           // 16*272*2 =  8704
#define OFF_YS    28160           // 4*16*4
#define OFF_XS    28416
#define OFF_RR0   28672
#define OFF_RR1   28928
#define OFF_WGT   29184
#define OFF_CELL  29440           // 16*2*4 = 128
#define WAVE_BYTES 29696
#define SMEM_BYTES (4 * WAVE_BYTES)   // 118784 B / block (4 waves)

#define SLAB_HALVES (16 * 512)    // one ktile x 16 ntiles of fragments

// ---------------------------------------------------------------------------
// Prep kernel 1: feat [B][C][H][W] f32  ->  featT [B][H][W][C] f16
// ---------------------------------------------------------------------------
__global__ void liif_transpose_feat(const float* __restrict__ feat,
                                    _Float16* __restrict__ featT, int n) {
  int i = blockIdx.x * 256 + threadIdx.x;
  if (i >= n) return;
  int c = i & 63, x = (i >> 6) & 63, y = (i >> 12) & 63, b = i >> 18;
  featT[i] = (_Float16)feat[(((b * LIIF_C + c) * LIIF_H + y) * LIIF_W) + x];
}

// ---------------------------------------------------------------------------
// Prep kernel 2: pack f32 weights into f16 WMMA B-fragment order.
// Fragment (kt,nt): 32 lanes x 16 halves, lane-contiguous (32B/lane).
// B lane layout packed here (contiguous K, same order as A): N = lane&15,
//   K = kt*32 + 16*(lane>=16) + h   (h = half index 0..15).
// layer0: GEMM K-index is permuted tap-major: K<576 -> orig row (K&63)*9+(K>>6)
// (The arrays are over-allocated by one pad slab; pad is loaded by the
//  software pipeline's tail refills but never consumed by a WMMA.)
// ---------------------------------------------------------------------------
__global__ void liif_pack_weights(const float* __restrict__ W,
                                  _Float16* __restrict__ dst,
                                  int ntiles, int Korig, int Ncols,
                                  int layer0, int n) {
  int i = blockIdx.x * 256 + threadIdx.x;
  if (i >= n) return;
  int h    = i & 15;
  int lane = (i >> 4) & 31;
  int frag = i >> 9;
  int nt   = frag % ntiles;
  int kt   = frag / ntiles;
  int K    = kt * 32 + ((lane & 16) ? 16 : 0) + h;
  int col  = nt * 16 + (lane & 15);
  float v = 0.0f;
  if (col < Ncols) {
    int row = -1;
    if (layer0) {
      if (K < 576)      row = (K & 63) * 9 + (K >> 6);   // tap-major permute
      else if (K < 580) row = K;                          // r0, r1, cell0, cell1
    } else if (K < Korig) {
      row = K;
    }
    if (row >= 0) v = W[row * Ncols + col];
  }
  dst[i] = (_Float16)v;
}

// ---------------------------------------------------------------------------
// Hidden layer: 16xK (LDS f16) @ KxN256 (global packed f16) -> relu -> LDS f16
// ktile-outer / ntile-inner keeps 16 v8f accumulators live; B fragments flow
// through a 4-deep rotating register pipeline carried across ktile boundaries.
// ---------------------------------------------------------------------------
template <int KT>
__device__ __forceinline__ void mlp_hidden(const _Float16* __restrict__ inb, int instride,
                                           const _Float16* __restrict__ wp,
                                           const float* __restrict__ bias,
                                           _Float16* __restrict__ outb, int outstride,
                                           int lane) {
  v8f acc[16] = {};
  const int mrow = lane & 15;
  const int kh   = (lane & 16) ? 16 : 0;  // A layout: lanes>=16 hold K+16 block
  const _Float16* wk = wp + lane * 16;

  // prime the 4-deep B pipeline
  v16h bf[4];
#pragma unroll
  for (int i = 0; i < 4; ++i) bf[i] = *(const v16h*)(wk + i * 512);

#pragma unroll 1
  for (int kt = 0; kt < KT; ++kt) {
    const _Float16* ar = inb + mrow * instride + kt * 32;
    v8h alo = *(const v8h*)(ar + kh);          // K = kh .. kh+7
    v8h ahi = *(const v8h*)(ar + kh + 8);      // K = kh+8 .. kh+15
    v16h a  = __builtin_shufflevector(alo, ahi, 0, 1, 2, 3, 4, 5, 6, 7,
                                      8, 9, 10, 11, 12, 13, 14, 15);
#pragma unroll
    for (int nt = 0; nt < 16; ++nt) {
      v16h cur = bf[nt & 3];
      // refill: fragments nt+4 of this ktile, or nt-12 of the next ktile.
      // Unconditional — the weight arrays carry one pad slab past the end.
      bf[nt & 3] = (nt < 12) ? *(const v16h*)(wk + (nt + 4) * 512)
                             : *(const v16h*)(wk + SLAB_HALVES + (nt - 12) * 512);
      acc[nt] = __builtin_amdgcn_wmma_f32_16x16x32_f16(
          false, a, false, cur, (short)0, acc[nt], false, false);
    }
    wk += SLAB_HALVES;
  }

  // epilogue: bias + relu + store via C layout (M = r + 8*(lane>=16), N = lane&15)
  const int mb = (lane & 16) ? 8 : 0;
#pragma unroll
  for (int nt = 0; nt < 16; ++nt) {
    const int col = nt * 16 + mrow;
    const float bv = bias[col];
#pragma unroll
    for (int r = 0; r < 8; ++r) {
      float v = acc[nt][r] + bv;
      v = v > 0.0f ? v : 0.0f;
      outb[(mb + r) * outstride + col] = (_Float16)v;
    }
  }
}

// ---------------------------------------------------------------------------
// Main kernel: 128 threads = 4 waves; each wave owns 16 queries of one batch,
// loops the 4 ensemble shifts, accumulates area-weighted head outputs in regs.
// ---------------------------------------------------------------------------
__global__ void __launch_bounds__(128)
liif_main(const _Float16* __restrict__ featT,
          const _Float16* __restrict__ w0p, const _Float16* __restrict__ w1p,
          const _Float16* __restrict__ w2p, const _Float16* __restrict__ w3p,
          const _Float16* __restrict__ w4p,
          const float* __restrict__ coord, const float* __restrict__ cell,
          const float* __restrict__ b0, const float* __restrict__ b1,
          const float* __restrict__ b2, const float* __restrict__ b3,
          const float* __restrict__ b4,
          float* __restrict__ out) {
  extern __shared__ char smem[];
  const int lane = threadIdx.x & 31;
  const int wave = threadIdx.x >> 5;
  char* wb = smem + wave * WAVE_BYTES;
  _Float16* buf0 = (_Float16*)(wb + OFF_BUF0);
  _Float16* buf1 = (_Float16*)(wb + OFF_BUF1);
  int*   ysL  = (int*)(wb + OFF_YS);
  int*   xsL  = (int*)(wb + OFF_XS);
  float* rr0L = (float*)(wb + OFF_RR0);
  float* rr1L = (float*)(wb + OFF_RR1);
  float* wgtL = (float*)(wb + OFF_WGT);
  float* celL = (float*)(wb + OFF_CELL);

  const int tile = blockIdx.x * 4 + wave;   // 16384 tiles total
  const int bb   = tile >> 12;              // / (Q/16)
  const int q0   = (tile & 4095) << 4;

  // ---- Phase A: per-query sample geometry for all 4 ensemble shifts --------
  if (lane < 16) {
    const int m = lane;
    const long qi = (long)bb * LIIF_Q + q0 + m;
    const float c0i = coord[qi * 2 + 0];
    const float c1i = coord[qi * 2 + 1];
    celL[m * 2 + 0] = cell[qi * 2 + 0] * (float)LIIF_H;   // rel_cell
    celL[m * 2 + 1] = cell[qi * 2 + 1] * (float)LIIF_W;
    float areas[4];
#pragma unroll
    for (int s = 0; s < 4; ++s) {
      const float vx = (s & 2) ? 1.0f : -1.0f;   // vx outer in reference
      const float vy = (s & 1) ? 1.0f : -1.0f;
      const float rx = 1.0f / 64.0f;
      float a0 = c0i + vx * rx + 1e-6f;
      float a1 = c1i + vy * rx + 1e-6f;
      a0 = fminf(fmaxf(a0, -1.0f + 1e-6f), 1.0f - 1e-6f);
      a1 = fminf(fmaxf(a1, -1.0f + 1e-6f), 1.0f - 1e-6f);
      int iy = (int)rintf((a0 + 1.0f) * 32.0f - 0.5f);   // RTNE = round-half-even
      int ix = (int)rintf((a1 + 1.0f) * 32.0f - 0.5f);
      iy = iy < 0 ? 0 : (iy > 63 ? 63 : iy);
      ix = ix < 0 ? 0 : (ix > 63 ? 63 : ix);
      const float qc0 = -1.0f + (2.0f * (float)iy + 1.0f) / 64.0f;
      const float qc1 = -1.0f + (2.0f * (float)ix + 1.0f) / 64.0f;
      const float r0 = (c0i - qc0) * 64.0f;
      const float r1 = (c1i - qc1) * 64.0f;
      ysL[s * 16 + m] = iy;  xsL[s * 16 + m] = ix;
      rr0L[s * 16 + m] = r0; rr1L[s * 16 + m] = r1;
      areas[s] = fabsf(r0 * r1) + 1e-9f;
    }
    const float tot = areas[0] + areas[1] + areas[2] + areas[3];
#pragma unroll
    for (int s = 0; s < 4; ++s) wgtL[s * 16 + m] = areas[3 - s] / tot;  // swap
  }
  __syncthreads();

  v8f outacc = {};   // lanes with (lane&15)<3 carry the 3 output channels

  for (int s = 0; s < 4; ++s) {
    // ---- Phase B: gather 16x608 A-panel into buf0 (tap-major K order) ------
    // 16 rows x 9 taps x 64ch = 1152 chunks of 8 halves (16B), 36/lane.
#pragma unroll 1
    for (int u = lane; u < 1152; u += 32) {
      const int m = u / 72;
      const int v = u - m * 72;
      const int t = v >> 3;
      const int p = v & 7;
      const int iy = ysL[s * 16 + m] + (t / 3) - 1;
      const int ix = xsL[s * 16 + m] + (t % 3) - 1;
      v8h val = {};
      if ((unsigned)iy < 64u && (unsigned)ix < 64u)
        val = *(const v8h*)(featT + ((((bb * 64) + iy) * 64 + ix) * 64) + p * 8);
      *(v8h*)(buf0 + m * STR0 + t * 64 + p * 8) = val;
    }
    if (lane < 16) {   // K tail: r0, r1, rel_cell, zero pad to 608
      const int m = lane;
      _Float16* row = buf0 + m * STR0;
      row[576] = (_Float16)rr0L[s * 16 + m];
      row[577] = (_Float16)rr1L[s * 16 + m];
      row[578] = (_Float16)celL[m * 2 + 0];
      row[579] = (_Float16)celL[m * 2 + 1];
#pragma unroll
      for (int k = 580; k < 608; ++k) row[k] = (_Float16)0.0f;
    }
    __syncthreads();

    // ---- Phase C: MLP (4 hidden WMMA layers, ping-pong LDS buffers) --------
    mlp_hidden<19>(buf0, STR0, w0p, b0, buf1, STR1, lane);
    __syncthreads();
    mlp_hidden<8>(buf1, STR1, w1p, b1, buf0, STR0, lane);
    __syncthreads();
    mlp_hidden<8>(buf0, STR0, w2p, b2, buf1, STR1, lane);
    __syncthreads();
    mlp_hidden<8>(buf1, STR1, w3p, b3, buf0, STR0, lane);
    __syncthreads();

    // ---- head: 256 -> 3 (padded to N=16), weight by swapped area ratio -----
    {
      v8f hacc = {};
      const int mrow = lane & 15;
      const int kh = (lane & 16) ? 16 : 0;
      const _Float16* w4l = w4p + lane * 16;
      v16h bh = *(const v16h*)(w4l);          // 2-deep pipeline (1 pad frag)
#pragma unroll 1
      for (int kt = 0; kt < 8; ++kt) {
        v16h cur = bh;
        bh = *(const v16h*)(w4l + (kt + 1) * 512);
        const _Float16* ar = buf0 + mrow * STR0 + kt * 32;
        v8h alo = *(const v8h*)(ar + kh);
        v8h ahi = *(const v8h*)(ar + kh + 8);
        v16h a  = __builtin_shufflevector(alo, ahi, 0, 1, 2, 3, 4, 5, 6, 7,
                                          8, 9, 10, 11, 12, 13, 14, 15);
        hacc = __builtin_amdgcn_wmma_f32_16x16x32_f16(
            false, a, false, cur, (short)0, hacc, false, false);
      }
      const int j  = lane & 15;
      const int mb = (lane & 16) ? 8 : 0;
      const float bj = (j < 3) ? b4[j] : 0.0f;
#pragma unroll
      for (int r = 0; r < 8; ++r) {
        const float p = hacc[r] + bj;
        outacc[r] += p * wgtL[s * 16 + mb + r];
      }
    }
    __syncthreads();   // before next sample overwrites buf0
  }

  // ---- write blended output [B,Q,3] ----------------------------------------
  const int j = lane & 15;
  if (j < 3) {
    const int mb = (lane & 16) ? 8 : 0;
#pragma unroll
    for (int r = 0; r < 8; ++r) {
      const long qi = (long)bb * LIIF_Q + q0 + mb + r;
      out[qi * 3 + j] = outacc[r];
    }
  }
}

// ---------------------------------------------------------------------------
// Host launcher. ws layout (f16), each weight array gets one pad slab:
//   featT 2MB | W0p (19+1) slabs | W1..3p (8+1) slabs | W4p (8+1) frags
// ---------------------------------------------------------------------------
extern "C" void kernel_launch(void* const* d_in, const int* in_sizes, int n_in,
                              void* d_out, int out_size, void* d_ws, size_t ws_size,
                              hipStream_t stream) {
  (void)in_sizes; (void)n_in; (void)out_size; (void)ws_size;
  const float* feat  = (const float*)d_in[0];
  const float* coord = (const float*)d_in[1];
  const float* cell  = (const float*)d_in[2];
  const float* W0 = (const float*)d_in[3];  const float* b0 = (const float*)d_in[4];
  const float* W1 = (const float*)d_in[5];  const float* b1 = (const float*)d_in[6];
  const float* W2 = (const float*)d_in[7];  const float* b2 = (const float*)d_in[8];
  const float* W3 = (const float*)d_in[9];  const float* b3 = (const float*)d_in[10];
  const float* W4 = (const float*)d_in[11]; const float* b4 = (const float*)d_in[12];
  float* out = (float*)d_out;

  char* ws = (char*)d_ws;
  _Float16* featT = (_Float16*)(ws + 0);         // 1,048,576 halves = 2,097,152 B
  _Float16* w0p   = (_Float16*)(ws + 2097152);   // (19+1)*16*512 halves = 327,680 B
  _Float16* w1p   = (_Float16*)(ws + 2424832);   // ( 8+1)*16*512 halves = 147,456 B
  _Float16* w2p   = (_Float16*)(ws + 2572288);   // 147,456 B
  _Float16* w3p   = (_Float16*)(ws + 2719744);   // 147,456 B
  _Float16* w4p   = (_Float16*)(ws + 2867200);   // ( 8+1)*512 halves = 9,216 B

  liif_transpose_feat<<<4096, 256, 0, stream>>>(feat, featT, 1048576);
  liif_pack_weights<<<608, 256, 0, stream>>>(W0, w0p, 16, 580, 256, 1, 155648);
  liif_pack_weights<<<256, 256, 0, stream>>>(W1, w1p, 16, 256, 256, 0, 65536);
  liif_pack_weights<<<256, 256, 0, stream>>>(W2, w2p, 16, 256, 256, 0, 65536);
  liif_pack_weights<<<256, 256, 0, stream>>>(W3, w3p, 16, 256, 256, 0, 65536);
  liif_pack_weights<<<16, 256, 0, stream>>>(W4, w4p, 1, 256, 3, 0, 4096);

  // 16384 wave-tiles / 4 waves per block
  liif_main<<<4096, 128, SMEM_BYTES, stream>>>(featT, w0p, w1p, w2p, w3p, w4p,
                                               coord, cell, b0, b1, b2, b3, b4, out);
}
